// ConditionalDLFactorized19_74680891343541
// MI455X (gfx1250) — matrix operand order repaired and checked by reading, not compiled, not run on hardware
//
#include <hip/hip_runtime.h>
#include <math.h>

// Problem constants (match reference)
#define TT 1024
#define BB 32
#define CC 1024
#define KK 1024
#define EE 16

typedef _Float16 half4_t __attribute__((ext_vector_type(4)));
typedef _Float16 half8_t __attribute__((ext_vector_type(8)));
typedef _Float16 v16h    __attribute__((ext_vector_type(16)));
typedef float    v8f     __attribute__((ext_vector_type(8)));

// ---------------------------------------------------------------------------
// Kernel 1: routing (logits -> softmax -> resp) + importance loss scalar
// One block of 512 threads; thread t handles (b = t>>4, e = t&15).
// ---------------------------------------------------------------------------
__global__ void __launch_bounds__(512)
route_kernel(const float* __restrict__ key,      // [B][K]
             const float* __restrict__ aw,       // [K][E]
             const float* __restrict__ ab,       // [E]
             float* __restrict__ resp_out,       // [B][E]
             float* __restrict__ loss_out)       // scalar at d_out[T*B*C]
{
    __shared__ float slog[BB * EE];
    __shared__ float sresp[BB * EE];
    const int t = threadIdx.x;
    const int b = t >> 4;
    const int e = t & 15;

    float acc = ab[e];
    const float* kp = key + b * KK;
    for (int k = 0; k < KK; ++k)
        acc += kp[k] * aw[k * EE + e];
    slog[t] = acc;
    __syncthreads();

    if (t < BB) {
        float mx = -1e30f;
        for (int j = 0; j < EE; ++j) mx = fmaxf(mx, slog[t * EE + j]);
        float s = 0.0f;
        for (int j = 0; j < EE; ++j) {
            float ex = expf(slog[t * EE + j] - mx);
            sresp[t * EE + j] = ex;
            s += ex;
        }
        float inv = 1.0f / s;
        for (int j = 0; j < EE; ++j) sresp[t * EE + j] *= inv;
    }
    __syncthreads();

    resp_out[t] = sresp[t];

    if (t == 0) {
        float imp[EE];
        float mean = 0.0f;
        for (int j = 0; j < EE; ++j) {
            float s = 0.0f;
            for (int bb2 = 0; bb2 < BB; ++bb2) s += sresp[bb2 * EE + j];
            imp[j] = s;
            mean += s;
        }
        mean *= (1.0f / EE);
        float var = 0.0f;
        for (int j = 0; j < EE; ++j) {
            float d = imp[j] - mean;
            var += d * d;
        }
        var *= (1.0f / (EE - 1));           // torch.std is unbiased (ddof=1)
        *loss_out = 0.01f * sqrtf(var) / mean;
    }
}

// ---------------------------------------------------------------------------
// Kernel 2: convert x fp32 -> f16 (vectorized float4 -> half4)
// ---------------------------------------------------------------------------
__global__ void __launch_bounds__(256)
cvt_x_kernel(const float* __restrict__ x, _Float16* __restrict__ xh)
{
    const size_t idx = (size_t)blockIdx.x * blockDim.x + threadIdx.x; // one float4 chunk
    const float4 v = ((const float4*)x)[idx];
    half4_t h;
    h[0] = (_Float16)v.x; h[1] = (_Float16)v.y;
    h[2] = (_Float16)v.z; h[3] = (_Float16)v.w;
    ((half4_t*)xh)[idx] = h;
}

// ---------------------------------------------------------------------------
// Kernel 3: mix expert weights: w_f16[b][o][i] = sum_e resp[b][e]*pw[e][o*C+i]
// Each thread owns one float4 chunk of the C*C plane, reads 16 expert chunks
// once, produces all 32 batches (pw_w1 read exactly once from HBM).
// ---------------------------------------------------------------------------
__global__ void __launch_bounds__(256)
mix_w_kernel(const float* __restrict__ resp,   // [B][E]
             const float* __restrict__ pw,     // [E][C*C]
             _Float16* __restrict__ wh)        // [B][C*C]
{
    __shared__ float sresp[BB * EE];
    for (int i = threadIdx.x; i < BB * EE; i += blockDim.x) sresp[i] = resp[i];
    __syncthreads();

    const size_t plane = (size_t)CC * CC;           // 1M elements
    const size_t chunk = (size_t)blockIdx.x * blockDim.x + threadIdx.x; // float4 idx

    float4 v[EE];
    #pragma unroll
    for (int e = 0; e < EE; ++e)
        v[e] = ((const float4*)(pw + (size_t)e * plane))[chunk];

    for (int b = 0; b < BB; ++b) {
        float ax = 0.f, ay = 0.f, az = 0.f, aw_ = 0.f;
        #pragma unroll
        for (int e = 0; e < EE; ++e) {
            const float r = sresp[b * EE + e];
            ax += r * v[e].x; ay += r * v[e].y;
            az += r * v[e].z; aw_ += r * v[e].w;
        }
        half4_t h;
        h[0] = (_Float16)ax; h[1] = (_Float16)ay;
        h[2] = (_Float16)az; h[3] = (_Float16)aw_;
        ((half4_t*)(wh + (size_t)b * plane))[chunk] = h;
    }
}

// ---------------------------------------------------------------------------
// Kernel 4: main batched GEMM with WMMA + fused residual add.
// y[t,b,o] = sum_i w[b,o,i] * x[t,b,i];  out = x + y
// Block: 256 thr (8 waves), 128(M=t) x 128(N=o) tile; wave: 64x32 (4x2 wmma).
// Grid: (N/128, M/128, B) = (8, 8, 32).
// ---------------------------------------------------------------------------
__global__ void __launch_bounds__(256)
gemm_kernel(const _Float16* __restrict__ xh,   // [T][B][C] f16
            const _Float16* __restrict__ wh,   // [B][C][C] f16 (o-major, i contiguous)
            const float*    __restrict__ x,    // [T][B][C] f32
            float*          __restrict__ out)  // [T][B][C] f32
{
    const int b    = blockIdx.z;
    const int m0   = blockIdx.y * 128;           // t tile base
    const int n0   = blockIdx.x * 128;           // o tile base
    const int wv   = threadIdx.x >> 5;
    const int lane = threadIdx.x & 31;
    const int wm   = (wv & 1) * 64;              // wave M offset
    const int wn   = (wv >> 1) * 32;             // wave N offset
    const int lhi  = lane >> 4;                  // half-wave select
    const int llo  = lane & 15;

    const v8f zero = {0.f, 0.f, 0.f, 0.f, 0.f, 0.f, 0.f, 0.f};
    v8f acc[4][2];
    #pragma unroll
    for (int mi = 0; mi < 4; ++mi)
        #pragma unroll
        for (int ni = 0; ni < 2; ++ni)
            acc[mi][ni] = zero;

    // Per-lane base pointers (row-invariant across K)
    const _Float16* pa[4];
    #pragma unroll
    for (int mi = 0; mi < 4; ++mi) {
        const int t = m0 + wm + mi * 16 + llo;
        pa[mi] = xh + ((size_t)t * BB + b) * CC + lhi * 8;
    }
    const _Float16* pb[2];
    #pragma unroll
    for (int ni = 0; ni < 2; ++ni) {
        const int o = n0 + wn + ni * 16 + llo;
        pb[ni] = wh + ((size_t)b * CC + o) * CC + lhi * 8;
    }

    #pragma unroll 2
    for (int k0 = 0; k0 < CC; k0 += 32) {
        // A fragments: 16x32 f16 per ISA layout
        // lane L(0-15): M=L, elems0-7 = K(lhi*8..+7), elems8-15 = same+16
        v16h af[4];
        #pragma unroll
        for (int mi = 0; mi < 4; ++mi) {
            const _Float16* p = pa[mi] + k0;
            const half8_t lo = *(const half8_t*)(p);
            const half8_t hi = *(const half8_t*)(p + 16);
            af[mi] = __builtin_shufflevector(lo, hi,
                     0, 1, 2, 3, 4, 5, 6, 7, 8, 9, 10, 11, 12, 13, 14, 15);
            // Speculative prefetch of next K tile; harmless past end (dropped)
            __builtin_prefetch(p + 32, 0, 3);
        }
        // B fragments: 32x16 f16; lane L: N=L&15, elems0-7 = K(lhi*8..+7), elems8-15 = same+16
        v16h bf[2];
        #pragma unroll
        for (int ni = 0; ni < 2; ++ni) {
            const _Float16* p = pb[ni] + k0;
            const half8_t lo = *(const half8_t*)(p);
            const half8_t hi = *(const half8_t*)(p + 16);
            bf[ni] = __builtin_shufflevector(lo, hi,
                     0, 1, 2, 3, 4, 5, 6, 7, 8, 9, 10, 11, 12, 13, 14, 15);
            __builtin_prefetch(p + 32, 0, 3);
        }
        #pragma unroll
        for (int mi = 0; mi < 4; ++mi)
            #pragma unroll
            for (int ni = 0; ni < 2; ++ni)
                acc[mi][ni] = __builtin_amdgcn_wmma_f32_16x16x32_f16(
                    /*neg_a=*/false, af[mi], /*neg_b=*/false, bf[ni],
                    /*c_mod=*/(short)0, acc[mi][ni],
                    /*reuse_a=*/false, /*reuse_b=*/false);
    }

    // Epilogue: D lane L elem r -> (M = lhi*8 + r, N = llo); fuse residual add
    #pragma unroll
    for (int mi = 0; mi < 4; ++mi)
        #pragma unroll
        for (int ni = 0; ni < 2; ++ni)
            #pragma unroll
            for (int r = 0; r < 8; ++r) {
                const int t = m0 + wm + mi * 16 + lhi * 8 + r;
                const int o = n0 + wn + ni * 16 + llo;
                const size_t idx = ((size_t)t * BB + b) * CC + o;
                out[idx] = x[idx] + acc[mi][ni][r];
            }
}

// ---------------------------------------------------------------------------
extern "C" void kernel_launch(void* const* d_in, const int* in_sizes, int n_in,
                              void* d_out, int out_size, void* d_ws, size_t ws_size,
                              hipStream_t stream)
{
    const float* x   = (const float*)d_in[0];   // (T,B,C)
    const float* key = (const float*)d_in[1];   // (1,B,K)
    const float* aw  = (const float*)d_in[2];   // (K,E)
    const float* ab  = (const float*)d_in[3];   // (E,)
    const float* pw  = (const float*)d_in[4];   // (E,C*C)
    float* out = (float*)d_out;                 // T*B*C floats + 1 loss scalar

    const size_t N = (size_t)TT * BB * CC;      // 33,554,432

    // Workspace layout: resp (512 f32) | x_f16 (N halves) | w_f16 (B*C*C halves)
    float*     resp = (float*)d_ws;
    _Float16*  xh   = (_Float16*)((char*)d_ws + 4096);
    _Float16*  wh   = (_Float16*)((char*)d_ws + 4096 + N * sizeof(_Float16));

    // 1) routing + loss (loss goes to out[N])
    route_kernel<<<1, 512, 0, stream>>>(key, aw, ab, resp, out + N);

    // 2) x fp32 -> f16   (N/4 float4 chunks, 256 thr/blk)
    cvt_x_kernel<<<(unsigned)(N / 4 / 256), 256, 0, stream>>>(x, xh);

    // 3) mix expert weights -> f16   (C*C/4 chunks, 256 thr/blk)
    mix_w_kernel<<<(unsigned)((size_t)CC * CC / 4 / 256), 256, 0, stream>>>(resp, pw, wh);

    // 4) batched WMMA GEMM + residual
    gemm_kernel<<<dim3(CC / 128, TT / 128, BB), 256, 0, stream>>>(xh, wh, x, out);
}
